// LinearAttn_77970836292157
// MI455X (gfx1250) — matrix operand order repaired and checked by reading, not compiled, not run on hardware
//
#include <hip/hip_runtime.h>
#include <math.h>

typedef __bf16 bh16;
typedef __attribute__((ext_vector_type(16))) __bf16 v16bf;
typedef __attribute__((ext_vector_type(8)))  __bf16 v8bf;
typedef __attribute__((ext_vector_type(8)))  float  v8f;

#define EPSF 1e-6f
#define Bb 4
#define Nn 8192
#define Mm 1024
#define Dd 1024
#define Hh 16
#define DH 64
#define SEQ (Nn + Mm)

enum { MODE_QKV = 0, MODE_CKV = 1, MODE_ATTN = 2, MODE_PLAIN = 3 };

// ---------------------------------------------------------------- convert
__global__ void f32_to_bf16_kernel(const float* __restrict__ src,
                                   bh16* __restrict__ dst, long n) {
  long i = (long)blockIdx.x * blockDim.x + threadIdx.x;
  long stride = (long)gridDim.x * blockDim.x;
  for (; i < n; i += stride) dst[i] = (bh16)src[i];
}

// ---------------------------------------------------------------- generic WMMA GEMM
// C(128x64 tile) = A(M x K, row-major bf16) @ B(K x N, row-major bf16) [+ bias]
// 256 threads = 8 waves; wave w computes rows [w*16, w*16+16) x all 64 cols.
template <int MODE>
__global__ void __launch_bounds__(256)
gemm_kernel(const bh16* __restrict__ A, const bh16* __restrict__ Bw,
            const float* __restrict__ bias,
            int Mrows, int Kdim, int Ncols,
            float* __restrict__ outF,
            bh16* __restrict__ oq, bh16* __restrict__ ok, bh16* __restrict__ ov,
            const float* __restrict__ w0, const float* __restrict__ w1,
            const float* __restrict__ ksum, bh16* __restrict__ attnb,
            int rows_per_batch, int seq_off) {
  __shared__ bh16 Atile[128][32];
  __shared__ bh16 Bt[64][32];      // Bt[n][k] = B[k][n]
  __shared__ float Cst[128][64];
  __shared__ float ksumLds[64];

  const int t = threadIdx.x;
  const int lane = t & 31;
  const int wave = t >> 5;
  const int halfsel = lane >> 4;   // 0: lanes 0-15, 1: lanes 16-31
  const int lr = lane & 15;

  const int nBase = blockIdx.x * 64;
  const int mBase = blockIdx.y * 128;
  const int bh = blockIdx.z;

  const bh16* Ab = A;
  const bh16* Bb_ = Bw;
  if (MODE == MODE_ATTN) {
    Ab += (size_t)bh * Mrows * Kdim;
    Bb_ += (size_t)bh * Kdim * Ncols;
    if (t < 64) ksumLds[t] = ksum[bh * 64 + t];
  }

  v8f acc[4];
#pragma unroll
  for (int j = 0; j < 4; j++)
#pragma unroll
    for (int r = 0; r < 8; r++) acc[j][r] = 0.0f;

  const int ar = t >> 1;           // A-tile row this thread loads
  const int ac = (t & 1) * 16;     // A-tile col base
  const int bk = t >> 3;           // B-tile k row (0..31)
  const int bn = (t & 7) * 8;      // B-tile n base

  for (int k0 = 0; k0 < Kdim; k0 += 32) {
    {  // A tile: 128x32, row-major, 16 halves/thread
      const bh16* src = Ab + (size_t)(mBase + ar) * Kdim + k0 + ac;
      *(v8bf*)&Atile[ar][ac] = *(const v8bf*)(src);
      *(v8bf*)&Atile[ar][ac + 8] = *(const v8bf*)(src + 8);
    }
    {  // B tile: 32x64 -> transposed into Bt
      const bh16* src = Bb_ + (size_t)(k0 + bk) * Ncols + nBase + bn;
      v8bf tmp = *(const v8bf*)src;
#pragma unroll
      for (int i = 0; i < 8; i++) Bt[bn + i][bk] = tmp[i];
    }
    __syncthreads();

    // A fragment per ISA 16-bit A layout (16x32): lane<16 -> K{0..7,16..23}, lane>=16 -> K{8..15,24..31}
    const int m0 = wave * 16;
    const int arow = m0 + lr;
    const int ko = halfsel ? 8 : 0;
    v8bf alo = *(const v8bf*)&Atile[arow][ko];
    v8bf ahi = *(const v8bf*)&Atile[arow][ko + 16];
    v16bf afrag = __builtin_shufflevector(alo, ahi, 0, 1, 2, 3, 4, 5, 6, 7, 8,
                                          9, 10, 11, 12, 13, 14, 15);
    const int kob = halfsel ? 8 : 0;  // B layout: lanes0-15 K{0..7,16..23}, lanes16-31 K{8..15,24..31}
#pragma unroll
    for (int j = 0; j < 4; j++) {
      const int n = j * 16 + lr;
      v8bf blo = *(const v8bf*)&Bt[n][kob];
      v8bf bhi = *(const v8bf*)&Bt[n][kob + 16];
      v16bf bfrag = __builtin_shufflevector(blo, bhi, 0, 1, 2, 3, 4, 5, 6, 7,
                                            8, 9, 10, 11, 12, 13, 14, 15);
      acc[j] = __builtin_amdgcn_wmma_f32_16x16x32_bf16(
          false, afrag, false, bfrag, (short)0, acc[j], false, false);
    }
    __syncthreads();
  }

  if (MODE != MODE_ATTN) {  // bias (b_qkv / b_ckv / b_out)
#pragma unroll
    for (int j = 0; j < 4; j++) {
      float bv = bias[nBase + j * 16 + lr];
#pragma unroll
      for (int r = 0; r < 8; r++) acc[j][r] += bv;
    }
  }

  const int m0 = wave * 16;
  if (MODE == MODE_PLAIN) {  // direct f32 store (final projection)
#pragma unroll
    for (int j = 0; j < 4; j++)
#pragma unroll
      for (int r = 0; r < 8; r++) {
        int row = mBase + m0 + r + halfsel * 8;  // C layout: VGPR r -> M=r / M=8+r
        int col = nBase + j * 16 + lr;
        outF[(size_t)row * Ncols + col] = acc[j][r];
      }
    return;
  }

  // stage C tile to LDS for row-wise epilogues (64 cols == one head)
#pragma unroll
  for (int j = 0; j < 4; j++)
#pragma unroll
    for (int r = 0; r < 8; r++)
      Cst[m0 + r + halfsel * 8][j * 16 + lr] = acc[j][r];
  __syncthreads();
  if (t >= 128) return;

  const int gRow = mBase + t;

  if (MODE == MODE_ATTN) {
    const int b = bh >> 4;
    const int h = bh & 15;
    const bh16* qrow = Ab + (size_t)gRow * 64;
    float scale = 0.0f;
#pragma unroll
    for (int d = 0; d < 64; d++) scale += (float)qrow[d] * ksumLds[d];
    const float invs = 1.0f / (scale + EPSF);
    bh16* dst = attnb + ((size_t)b * Nn + gRow) * Dd + h * 64;
#pragma unroll
    for (int e = 0; e < 64; e++) dst[e] = (bh16)(Cst[t][e] * invs);
    return;
  }

  // MODE_QKV: sections q|k|v at col/1024; MODE_CKV: sections k|v
  const int sec = nBase >> 10;
  const int h = (nBase & 1023) >> 6;
  const int b = gRow / rows_per_batch;
  const int n = gRow - b * rows_per_batch;

  const bool isQ = (MODE == MODE_QKV) && (sec == 0);
  const bool isK = (MODE == MODE_QKV) ? (sec == 1) : (sec == 0);

  if (!isQ && !isK) {  // V: plain copy to (B,H,SEQ,DH)
    bh16* dst = ov + (((size_t)(b * Hh + h)) * SEQ + seq_off + n) * 64;
#pragma unroll
    for (int d = 0; d < 64; d++) dst[d] = (bh16)Cst[t][d];
    return;
  }
  float ssq = 0.0f;
#pragma unroll
  for (int d = 0; d < 64; d++) {
    float v = Cst[t][d];
    ssq += v * v;
  }
  const float inv = rsqrtf(ssq * (1.0f / 64.0f) + EPSF);
  const float* w = ((MODE == MODE_QKV) && (sec == 1)) ? w1 : w0;
  bh16* dst = isQ ? (oq + (((size_t)(b * Hh + h)) * rows_per_batch + n) * 64)
                  : (ok + (((size_t)(b * Hh + h)) * SEQ + seq_off + n) * 64);
#pragma unroll
  for (int d = 0; d < 64; d++)
    dst[d] = (bh16)fmaxf(Cst[t][d] * inv * w[d], 0.0f);
}

// ---------------------------------------------------------------- k_sum over seq
__global__ void __launch_bounds__(256)
ksum_kernel(const bh16* __restrict__ kb, float* __restrict__ ksum) {
  __shared__ float red[256];
  const int bh = blockIdx.x;
  const int t = threadIdx.x;
  const int d = t & 63;
  const int c = t >> 6;
  const bh16* base = kb + (size_t)bh * SEQ * 64;
  float s = 0.0f;
  for (int seq = c; seq < SEQ; seq += 4) s += (float)base[(size_t)seq * 64 + d];
  red[t] = s;
  __syncthreads();
  if (t < 64) ksum[bh * 64 + t] = red[t] + red[t + 64] + red[t + 128] + red[t + 192];
}

// ---------------------------------------------------------------- kv = k^T v per (b,h)
__global__ void __launch_bounds__(256)
kv_kernel(const bh16* __restrict__ kb, const bh16* __restrict__ vb,
          bh16* __restrict__ kvb) {
  __shared__ bh16 kt[64][32];  // kt[d][kk] = k[s0+kk][d]
  __shared__ bh16 vt[64][32];  // vt[e][kk] = v[s0+kk][e]
  const int bh = blockIdx.x;
  const int t = threadIdx.x;
  const int lane = t & 31;
  const int wave = t >> 5;
  const int halfsel = lane >> 4;
  const int lr = lane & 15;
  const bh16* kbase = kb + (size_t)bh * SEQ * 64;
  const bh16* vbase = vb + (size_t)bh * SEQ * 64;
  const int dRow = t >> 2;        // 0..63
  const int kkB = (t & 3) * 8;    // 0,8,16,24

  v8f acc[2];
#pragma unroll
  for (int j = 0; j < 2; j++)
#pragma unroll
    for (int r = 0; r < 8; r++) acc[j][r] = 0.0f;

  const int m0 = (wave >> 1) * 16;
  const int nB = (wave & 1) * 32;

  for (int s0 = 0; s0 < SEQ; s0 += 32) {
    v8bf tk, tv;
#pragma unroll
    for (int i = 0; i < 8; i++) {
      tk[i] = kbase[(size_t)(s0 + kkB + i) * 64 + dRow];
      tv[i] = vbase[(size_t)(s0 + kkB + i) * 64 + dRow];
    }
    *(v8bf*)&kt[dRow][kkB] = tk;
    *(v8bf*)&vt[dRow][kkB] = tv;
    __syncthreads();

    const int arow = m0 + lr;
    const int ko = halfsel ? 8 : 0;
    v8bf alo = *(const v8bf*)&kt[arow][ko];
    v8bf ahi = *(const v8bf*)&kt[arow][ko + 16];
    v16bf afrag = __builtin_shufflevector(alo, ahi, 0, 1, 2, 3, 4, 5, 6, 7, 8,
                                          9, 10, 11, 12, 13, 14, 15);
    const int kob = halfsel ? 8 : 0;
#pragma unroll
    for (int j = 0; j < 2; j++) {
      const int n = nB + j * 16 + lr;
      v8bf blo = *(const v8bf*)&vt[n][kob];
      v8bf bhi = *(const v8bf*)&vt[n][kob + 16];
      v16bf bfrag = __builtin_shufflevector(blo, bhi, 0, 1, 2, 3, 4, 5, 6, 7,
                                            8, 9, 10, 11, 12, 13, 14, 15);
      acc[j] = __builtin_amdgcn_wmma_f32_16x16x32_bf16(
          false, afrag, false, bfrag, (short)0, acc[j], false, false);
    }
    __syncthreads();
  }

  bh16* dst = kvb + (size_t)bh * 64 * 64;
#pragma unroll
  for (int j = 0; j < 2; j++)
#pragma unroll
    for (int r = 0; r < 8; r++) {
      int row = m0 + r + halfsel * 8;
      int col = nB + j * 16 + lr;
      dst[row * 64 + col] = (bh16)acc[j][r];
    }
}

// ---------------------------------------------------------------- launch
extern "C" void kernel_launch(void* const* d_in, const int* in_sizes, int n_in,
                              void* d_out, int out_size, void* d_ws,
                              size_t ws_size, hipStream_t stream) {
  const float* x = (const float*)d_in[0];
  const float* y = (const float*)d_in[1];
  const float* w_qkv = (const float*)d_in[2];
  const float* b_qkv = (const float*)d_in[3];
  const float* w_ckv = (const float*)d_in[4];
  const float* b_ckv = (const float*)d_in[5];
  const float* w_out = (const float*)d_in[6];
  const float* b_out = (const float*)d_in[7];
  const float* qn_w = (const float*)d_in[8];
  const float* kn_w = (const float*)d_in[9];
  const float* ckn_w = (const float*)d_in[10];

  char* ws = (char*)d_ws;
  size_t off = 0;
  auto alloc = [&](size_t bytes) {
    void* p = ws + off;
    off += (bytes + 255) & ~(size_t)255;
    return p;
  };
  bh16* xb = (bh16*)alloc((size_t)Bb * Nn * Dd * 2);
  bh16* yb = (bh16*)alloc((size_t)Bb * Mm * Dd * 2);
  bh16* wqkvb = (bh16*)alloc((size_t)Dd * 3 * Dd * 2);
  bh16* wckvb = (bh16*)alloc((size_t)Dd * 2 * Dd * 2);
  bh16* woutb = (bh16*)alloc((size_t)Dd * Dd * 2);
  bh16* qb = (bh16*)alloc((size_t)Bb * Hh * Nn * DH * 2);
  bh16* kbuf = (bh16*)alloc((size_t)Bb * Hh * SEQ * DH * 2);
  bh16* vbuf = (bh16*)alloc((size_t)Bb * Hh * SEQ * DH * 2);
  bh16* kvb = (bh16*)alloc((size_t)Bb * Hh * DH * DH * 2);
  float* ksum = (float*)alloc((size_t)Bb * Hh * DH * 4);
  bh16* attnb = (bh16*)alloc((size_t)Bb * Nn * Dd * 2);
  (void)ws_size;

  // 1) f32 -> bf16 conversions
  f32_to_bf16_kernel<<<4096, 256, 0, stream>>>(x, xb, (long)Bb * Nn * Dd);
  f32_to_bf16_kernel<<<2048, 256, 0, stream>>>(y, yb, (long)Bb * Mm * Dd);
  f32_to_bf16_kernel<<<2048, 256, 0, stream>>>(w_qkv, wqkvb, (long)Dd * 3 * Dd);
  f32_to_bf16_kernel<<<2048, 256, 0, stream>>>(w_ckv, wckvb, (long)Dd * 2 * Dd);
  f32_to_bf16_kernel<<<1024, 256, 0, stream>>>(w_out, woutb, (long)Dd * Dd);

  // 2) QKV projection fused with RMSNorm+ReLU, scatter into (B,H,seq,DH)
  gemm_kernel<MODE_QKV><<<dim3(3 * Dd / 64, (Bb * Nn) / 128, 1), 256, 0, stream>>>(
      xb, wqkvb, b_qkv, Bb * Nn, Dd, 3 * Dd, nullptr, qb, kbuf, vbuf, qn_w,
      kn_w, nullptr, nullptr, Nn, 0);

  // 3) cross-KV projection fused with RMSNorm+ReLU at seq offset N
  gemm_kernel<MODE_CKV><<<dim3(2 * Dd / 64, (Bb * Mm) / 128, 1), 256, 0, stream>>>(
      yb, wckvb, b_ckv, Bb * Mm, Dd, 2 * Dd, nullptr, nullptr, kbuf, vbuf,
      ckn_w, nullptr, nullptr, nullptr, Mm, Nn);

  // 4) k_sum and kv = k^T v per (b,h)
  ksum_kernel<<<Bb * Hh, 256, 0, stream>>>(kbuf, ksum);
  kv_kernel<<<Bb * Hh, 256, 0, stream>>>(kbuf, vbuf, kvb);

  // 5) attn = q @ kv, divided by (q . k_sum + eps), packed to (B,N,D) bf16
  gemm_kernel<MODE_ATTN><<<dim3(1, Nn / 128, Bb * Hh), 256, 0, stream>>>(
      qb, kvb, nullptr, Nn, DH, DH, nullptr, nullptr, nullptr, nullptr,
      nullptr, nullptr, ksum, attnb, Nn, 0);

  // 6) output projection -> f32 d_out
  gemm_kernel<MODE_PLAIN><<<dim3(Dd / 64, (Bb * Nn) / 128, 1), 256, 0, stream>>>(
      attnb, woutb, b_out, Bb * Nn, Dd, Dd, (float*)d_out, nullptr, nullptr,
      nullptr, nullptr, nullptr, nullptr, nullptr, Nn, 0);
}
